// OrchidOperator_41867341201683
// MI455X (gfx1250) — matrix operand order repaired and checked by reading, not compiled, not run on hardware
//
#include <hip/hip_runtime.h>
#include <math.h>

// ---------------------------------------------------------------------------
// Types for CDNA5 WMMA (wave32, 16x16x32 bf16 -> f32)
// ---------------------------------------------------------------------------
typedef __attribute__((ext_vector_type(16))) __bf16 v16bf;
typedef __attribute__((ext_vector_type(8)))  float  v8f;

__device__ __forceinline__ __bf16 f2bf(float f) {
    union { float f; unsigned u; } in;
    in.f = f;
    unsigned r = (in.u + 0x7FFFu + ((in.u >> 16) & 1u)) >> 16; // round-to-nearest-even
    union { unsigned short s; __bf16 b; } out;
    out.s = (unsigned short)r;
    return out.b;
}

__device__ __forceinline__ float gelu_exact(float x) {
    return 0.5f * x * (1.0f + erff(x * 0.70710678118654752f));
}

// low 32 bits of a generic pointer to LDS == wave-relative LDS byte address
__device__ __forceinline__ unsigned lds_addr_of(const void* p) {
    return (unsigned)(unsigned long long)p;
}

// ---------------------------------------------------------------------------
// Generic GEMM:  C[z][m][n] = act( A[z][m][:] . W[n][:] + bias[n] )
//   A: row-major (lda), per-z stride a_bs.   W: row-major (N,K).
//   trans_store=0: C[m*ldc + n]   trans_store=1: C[n*ldc + m]  (channel-major)
//
// Block = 256 threads = 8 waves. Block tile: 128(M) x 64(N).
//   wave w owns m-tile (blockIdx.y*8 + w) and the full 64-col N strip:
//   one A fragment feeds 4 x v_wmma_f32_16x16x32_bf16 per K step.
// The 64x32 f32 W panel is staged per block into LDS with
// global_load_async_to_lds_b128 (ASYNCcnt) and gathered via ds_load_b128.
// Requires M%128==0, N%64==0, K%32==0 (true for all call sites).
// ---------------------------------------------------------------------------
__global__ void __launch_bounds__(256)
gemm_bf16_wmma(const float* __restrict__ A, int lda, long long a_bs,
               const float* __restrict__ W,
               const float* __restrict__ bias,
               float* __restrict__ C, long long c_bs, int ldc,
               int M, int N, int K, int act, int trans_store)
{
    __shared__ float wtile[64 * 32];   // 8 KB W panel: wtile[n_local][k_local]

    const int tid  = threadIdx.x;
    const int lane = tid & 31;
    const int wave = tid >> 5;
    const int m0   = (blockIdx.y * 8 + wave) * 16;   // 16-row tile per wave
    const int n0   = blockIdx.x * 64;                // 64-col strip per block

    const int half = lane >> 4;     // lane group 0: lanes 0-15, group 1: 16-31
    const int mn   = lane & 15;     // row (A) / col (B,C) index within tile

    const float* arow = A + (long long)blockIdx.z * a_bs + (long long)(m0 + mn) * lda;

    v8f acc[4];
#pragma unroll
    for (int j = 0; j < 4; ++j) {
        const float bv = bias ? bias[n0 + j * 16 + mn] : 0.0f;
#pragma unroll
        for (int r = 0; r < 8; ++r) acc[j][r] = bv;
    }

    // per-thread staging slots: thread t fills flat elements [t*8, t*8+8)
    const unsigned lds_base = lds_addr_of(wtile) + (unsigned)tid * 32u;

    for (int k0 = 0; k0 < K; k0 += 32) {
        __syncthreads();   // previous tile fully consumed
        // ---- async stage of W panel: 64 rows x 32 k (f32) ----
#pragma unroll
        for (int i = 0; i < 2; ++i) {
            const int f  = tid * 8 + i * 4;      // flat f32 index in panel
            const int nn = f >> 5;               // n_local
            const int kk = f & 31;               // k_local
            const float* g = W + (long long)(n0 + nn) * K + (k0 + kk);
            asm volatile("global_load_async_to_lds_b128 %0, %1, off"
                         :: "v"(lds_base + (unsigned)(i * 16)),
                            "v"((unsigned long long)g)
                         : "memory");
        }
        asm volatile("s_wait_asynccnt 0x0" ::: "memory");
        __syncthreads();   // panel visible to all waves

        // ---- A fragment (16x32 bf16), ISA 7.12.2 layout ----
        v16bf af;
#pragma unroll
        for (int i = 0; i < 8; ++i) {
            const int kk = (i < 4) ? (half * 8 + 2 * i) : (16 + half * 8 + 2 * (i - 4));
            float2 t = *(const float2*)(arow + k0 + kk);
            af[2 * i]     = f2bf(t.x);
            af[2 * i + 1] = f2bf(t.y);
        }
        if (k0 + 32 < K)
            __builtin_prefetch(arow + k0 + 32, 0, 0);   // global_prefetch_b8

        // ---- 4 x WMMA over the 64-col strip, A fragment reused ----
#pragma unroll
        for (int j = 0; j < 4; ++j) {
            // B fragment: lane col = j*16+mn; lanes 0-15 hold K=0..7,16..23, 16-31 hold 8..15,24..31
            const float* wl = wtile + (j * 16 + mn) * 32 + half * 8;
            v16bf bfr;
#pragma unroll
            for (int i = 0; i < 4; ++i) {
                float4 t = *(const float4*)(wl + 4 * i + (i >= 2 ? 8 : 0));   // ds_load_b128
                bfr[4 * i]     = f2bf(t.x);
                bfr[4 * i + 1] = f2bf(t.y);
                bfr[4 * i + 2] = f2bf(t.z);
                bfr[4 * i + 3] = f2bf(t.w);
            }
            acc[j] = __builtin_amdgcn_wmma_f32_16x16x32_bf16(
                /*neg_a=*/false, af, /*neg_b=*/false, bfr,
                /*c_mod=*/(short)0, acc[j], /*reuse_a=*/false, /*reuse_b=*/false);
        }
    }

    if (act == 1) {
#pragma unroll
        for (int j = 0; j < 4; ++j)
#pragma unroll
            for (int r = 0; r < 8; ++r) acc[j][r] = gelu_exact(acc[j][r]);
    }

    // C/D layout: lane column n; VGPR r holds row m = m0 + half*8 + r
    if (!trans_store) {
        float* cbase = C + (long long)blockIdx.z * c_bs;
#pragma unroll
        for (int j = 0; j < 4; ++j)
#pragma unroll
            for (int r = 0; r < 8; ++r)
                cbase[(long long)(m0 + half * 8 + r) * ldc + (n0 + j * 16 + mn)] = acc[j][r];
    } else {
#pragma unroll
        for (int j = 0; j < 4; ++j) {
            float* p = C + (long long)blockIdx.z * c_bs
                         + (long long)(n0 + j * 16 + mn) * ldc + (m0 + half * 8);
            float4 lo = { acc[j][0], acc[j][1], acc[j][2], acc[j][3] };
            float4 hi = { acc[j][4], acc[j][5], acc[j][6], acc[j][7] };
            *(float4*)(p)     = lo;
            *(float4*)(p + 4) = hi;
        }
    }
}

// ---------------------------------------------------------------------------
// Depthwise causal conv (w_sf, pad=2 then truncate -> taps at l-2, l-1, l)
// layout (B, C, L) contiguous
// ---------------------------------------------------------------------------
__global__ void __launch_bounds__(256)
dwconv_sf_k(const float* __restrict__ in, const float* __restrict__ w,
            const float* __restrict__ bias, float* __restrict__ out,
            int L, int C, int total)
{
    int idx = blockIdx.x * 256 + threadIdx.x;
    if (idx >= total) return;
    int l = idx % L;
    int c = (idx / L) % C;
    float s = bias[c] + w[c * 3 + 2] * in[idx];
    if (l >= 1) s += w[c * 3 + 1] * in[idx - 1];
    if (l >= 2) s += w[c * 3 + 0] * in[idx - 2];
    out[idx] = s;
}

// ---------------------------------------------------------------------------
// Depthwise centered conv (pad=1), optional ReLU.  in/out (B, C, L) strided.
// ---------------------------------------------------------------------------
__global__ void __launch_bounds__(256)
dwconv3_k(const float* __restrict__ in, long long in_bs,
          const float* __restrict__ w, const float* __restrict__ bias,
          float* __restrict__ out, long long out_bs,
          int C, int L, int relu_flag, int total)
{
    int idx = blockIdx.x * 256 + threadIdx.x;
    if (idx >= total) return;
    int l = idx % L;
    int c = (idx / L) % C;
    int b = idx / (L * C);
    const float* row = in + (long long)b * in_bs + (long long)c * L;
    float s = bias[c] + w[c * 3 + 1] * row[l];
    if (l > 0)     s += w[c * 3 + 0] * row[l - 1];
    if (l < L - 1) s += w[c * 3 + 2] * row[l + 1];
    if (relu_flag) s = fmaxf(s, 0.0f);
    out[(long long)b * out_bs + (long long)c * L + l] = s;
}

// ---------------------------------------------------------------------------
// h_adapt: per channel c, mean_b |FFT_8192(h[b,c,:])|, then inverse transform.
// |H| of real input is real & Hermitian-symmetric, so irfft == forward FFT / N.
// Radix-2 in-place DIT in LDS (bit-reversed load). 256 threads, 96 KB LDS.
// ---------------------------------------------------------------------------
__device__ __forceinline__ void fft8192_lds(float* re, float* im, int tid)
{
#pragma unroll 1
    for (int s = 1; s <= 13; ++s) {
        const int hm = 1 << (s - 1);
        const int m  = hm << 1;
        __syncthreads();
        for (int j = tid; j < 4096; j += 256) {
            const int pos = j & (hm - 1);
            const int grp = j >> (s - 1);
            const int i0 = grp * m + pos;
            const int i1 = i0 + hm;
            const float ang = -6.28318530717958648f * (float)pos / (float)m;
            float sn, cs;
            __sincosf(ang, &sn, &cs);          // w = cs + i*sn  (e^{-i theta})
            const float tr = re[i1] * cs - im[i1] * sn;
            const float ti = re[i1] * sn + im[i1] * cs;
            re[i1] = re[i0] - tr;  im[i1] = im[i0] - ti;
            re[i0] = re[i0] + tr;  im[i0] = im[i0] + ti;
        }
    }
    __syncthreads();
}

__global__ void __launch_bounds__(256)
fft_hadapt_k(const float* __restrict__ h, float* __restrict__ h_adapt, int B)
{
    extern __shared__ float smem[];
    float* re  = smem;
    float* im  = smem + 8192;
    float* acc = smem + 16384;
    const int tid = threadIdx.x;
    const int c   = blockIdx.x;

    for (int i = tid; i < 8192; i += 256) acc[i] = 0.0f;

    for (int b = 0; b < B; ++b) {
        __syncthreads();
        const float* src = h + ((long long)b * 256 + c) * 8192;
        for (int i = tid; i < 8192; i += 256) {
            const int r = (int)(__brev((unsigned)i) >> 19);  // 13-bit reversal
            re[r] = src[i];
            im[r] = 0.0f;
        }
        fft8192_lds(re, im, tid);
        for (int i = tid; i < 8192; i += 256)
            acc[i] += sqrtf(re[i] * re[i] + im[i] * im[i]);
    }
    __syncthreads();

    // inverse: forward FFT of (real, symmetric) mean spectrum, scale by 1/(B*N)
    for (int i = tid; i < 8192; i += 256) {
        const int r = (int)(__brev((unsigned)i) >> 19);
        re[r] = acc[i];
        im[r] = 0.0f;
    }
    fft8192_lds(re, im, tid);
    const float scale = 1.0f / ((float)B * 8192.0f);
    for (int i = tid; i < 8192; i += 256)
        h_adapt[(long long)c * 8192 + i] = re[i] * scale;
}

// ---------------------------------------------------------------------------
// MLP layer 0 (K=3): h0a[l,c] = gelu(z . W0[c] + b0[c])
// ---------------------------------------------------------------------------
__global__ void __launch_bounds__(256)
mlp0_k(const float* __restrict__ z, const float* __restrict__ W0,
       const float* __restrict__ b0, float* __restrict__ out, int total)
{
    int idx = blockIdx.x * 256 + threadIdx.x;
    if (idx >= total) return;
    int c = idx % 768;
    int l = idx / 768;
    float s = b0[c] + z[l * 3 + 0] * W0[c * 3 + 0]
                    + z[l * 3 + 1] * W0[c * 3 + 1]
                    + z[l * 3 + 2] * W0[c * 3 + 2];
    out[idx] = gelu_exact(s);
}

// ---------------------------------------------------------------------------
// wfil[l,d] = h03[l,d] * exp(-t_pe[l]*|deltas[d]|) + h_adapt[d,l]
// ---------------------------------------------------------------------------
__global__ void __launch_bounds__(256)
combine_k(const float* __restrict__ h03, const float* __restrict__ t_pe,
          const float* __restrict__ deltas, const float* __restrict__ h_adapt,
          float* __restrict__ wfil, int total)
{
    int idx = blockIdx.x * 256 + threadIdx.x;
    if (idx >= total) return;
    int dd = idx & 255;
    int l  = idx >> 8;
    float dec = __expf(-t_pe[l] * fabsf(deltas[dd]));
    wfil[idx] = h03[idx] * dec + h_adapt[(long long)dd * 8192 + l];
}

// ---------------------------------------------------------------------------
// Circular convolution along d=256 (== irfft(rfft(a)*rfft(w)) over last axis):
//   a[m] = v*x0 at (b, m, l);  ynew[b,l,n] = (sum_m a[m]*w[(n-m)&255]) * x1[n]
// One 256-thread block per (l, b). Filter row staged into LDS with
// global_load_async_to_lds_b32 (ASYNCcnt); product row staged via ds stores.
// ---------------------------------------------------------------------------
__global__ void __launch_bounds__(256)
circconv_k(const float* __restrict__ uc, const float* __restrict__ wfil,
           float* __restrict__ ynew)
{
    __shared__ float sa[256];
    __shared__ float sw[256];
    const int n = threadIdx.x;
    const int l = blockIdx.x;
    const int b = blockIdx.y;

    // async gather of the filter row into LDS
    asm volatile("global_load_async_to_lds_b32 %0, %1, off"
                 :: "v"(lds_addr_of(&sw[n])),
                    "v"((unsigned long long)(wfil + l * 256 + n))
                 : "memory");

    const long long base = (long long)b * 768 * 8192 + l;
    const float x0v = uc[base + (long long)(n)       * 8192];
    const float x1v = uc[base + (long long)(256 + n) * 8192];
    const float vv  = uc[base + (long long)(512 + n) * 8192];
    sa[n] = vv * x0v;

    asm volatile("s_wait_asynccnt 0x0" ::: "memory");
    __syncthreads();

    float s = 0.0f;
#pragma unroll 8
    for (int m = 0; m < 256; ++m)
        s += sa[m] * sw[(n - m) & 255];
    ynew[((long long)b * 8192 + l) * 256 + n] = s * x1v;
}

// ---------------------------------------------------------------------------
// Host launch
// ---------------------------------------------------------------------------
extern "C" void kernel_launch(void* const* d_in, const int* in_sizes, int n_in,
                              void* d_out, int out_size, void* d_ws, size_t ws_size,
                              hipStream_t stream)
{
    (void)in_sizes; (void)n_in; (void)out_size; (void)ws_size;
    const float* u      = (const float*)d_in[0];
    const float* W_in   = (const float*)d_in[1];
    const float* b_in   = (const float*)d_in[2];
    const float* w_sf   = (const float*)d_in[3];
    const float* b_sf   = (const float*)d_in[4];
    const float* w_c1   = (const float*)d_in[5];
    const float* b_c1   = (const float*)d_in[6];
    const float* w_c2   = (const float*)d_in[7];
    const float* b_c2   = (const float*)d_in[8];
    const float* W0     = (const float*)d_in[9];
    const float* b0     = (const float*)d_in[10];
    const float* W1     = (const float*)d_in[11];
    const float* b1     = (const float*)d_in[12];
    const float* W2     = (const float*)d_in[13];
    const float* b2     = (const float*)d_in[14];
    const float* W3     = (const float*)d_in[15];
    const float* b3     = (const float*)d_in[16];
    const float* deltas = (const float*)d_in[17];
    const float* z_pe   = (const float*)d_in[18];
    const float* t_pe   = (const float*)d_in[19];
    const float* W_out  = (const float*)d_in[20];
    const float* b_out  = (const float*)d_in[21];
    float* out = (float*)d_out;

    const int B = 8, L = 8192, D = 256, IW = 768;

    // workspace layout (floats)
    float* ws    = (float*)d_ws;
    float* up    = ws;                                  // (B,768,L)
    float* uc    = up   + (size_t)B * IW * L;           // (B,768,L)
    float* h1    = uc   + (size_t)B * IW * L;           // (B,256,L)
    float* hbuf  = h1   + (size_t)B * D * L;            // (B,256,L)
    float* hadap = hbuf + (size_t)B * D * L;            // (256,L)
    float* h0a   = hadap+ (size_t)D * L;                // (L,768)
    float* h0b   = h0a  + (size_t)L * IW;               // (L,768)
    float* h03   = h0b  + (size_t)L * IW;               // (L,256)
    float* wfil  = h03  + (size_t)L * D;                // (L,256)
    float* ynew  = wfil + (size_t)L * D;                // (B,L,256)

    // 1) up = u @ W_in^T + b_in  -> channel-major (B,768,L)
    gemm_bf16_wmma<<<dim3(IW / 64, L / 128, B), 256, 0, stream>>>(
        u, D, (long long)L * D, W_in, b_in,
        up, (long long)IW * L, L, L, IW, D, /*act*/0, /*trans*/1);

    // 2) uc = causal depthwise conv (w_sf)
    {
        int total = B * IW * L;
        dwconv_sf_k<<<total / 256, 256, 0, stream>>>(up, w_sf, b_sf, uc, L, IW, total);
    }

    // 3) h1 = relu(dwconv(v, w_c1)); v = uc channels [512,768)
    {
        int total = B * D * L;
        dwconv3_k<<<total / 256, 256, 0, stream>>>(
            uc + (size_t)2 * D * L, (long long)IW * L, w_c1, b_c1,
            h1, (long long)D * L, D, L, /*relu*/1, total);
        // 4) h = dwconv(h1, w_c2)
        dwconv3_k<<<total / 256, 256, 0, stream>>>(
            h1, (long long)D * L, w_c2, b_c2,
            hbuf, (long long)D * L, D, L, /*relu*/0, total);
    }

    // 5) h_adapt = irfft(mean_b |rfft_8192(h)|)
    fft_hadapt_k<<<D, 256, 96 * 1024, stream>>>(hbuf, hadap, B);

    // 6) MLP layer 0 (K=3, gelu)
    {
        int total = L * IW;
        mlp0_k<<<total / 256, 256, 0, stream>>>(z_pe, W0, b0, h0a, total);
    }
    // 7,8) hidden layers 768x768 (gelu)
    gemm_bf16_wmma<<<dim3(IW / 64, L / 128, 1), 256, 0, stream>>>(
        h0a, IW, 0, W1, b1, h0b, 0, IW, L, IW, IW, /*act*/1, 0);
    gemm_bf16_wmma<<<dim3(IW / 64, L / 128, 1), 256, 0, stream>>>(
        h0b, IW, 0, W2, b2, h0a, 0, IW, L, IW, IW, /*act*/1, 0);
    // 9) output layer 768->256 (no act)
    gemm_bf16_wmma<<<dim3(D / 64, L / 128, 1), 256, 0, stream>>>(
        h0a, IW, 0, W3, b3, h03, 0, D, L, D, IW, /*act*/0, 0);

    // 10) wfil = h03 * exp(-t|delta|) + h_adapt^T
    {
        int total = L * D;
        combine_k<<<total / 256, 256, 0, stream>>>(h03, t_pe, deltas, hadap, wfil, total);
    }

    // 11) circular conv along d=256, times x1
    circconv_k<<<dim3(L, B, 1), 256, 0, stream>>>(uc, wfil, ynew);

    // 12) out = ynew @ W_out^T + b_out
    gemm_bf16_wmma<<<dim3(D / 64, (B * L) / 128, 1), 256, 0, stream>>>(
        ynew, D, 0, W_out, b_out, out, 0, D, B * L, D, D, /*act*/0, 0);
}
